// FeedbackAttention_66322884985016
// MI455X (gfx1250) — compile-verified, matrix operand order checked by reading, not run on hardware
//
#include <hip/hip_runtime.h>

#define B_ 16
#define H_ 16
#define DK 64
#define DM 1024
#define S_ 4096
#define BH (B_*H_)     // 256
#define BD (B_*DM)     // 16384

typedef __attribute__((ext_vector_type(16))) __bf16 v16bf;
typedef __attribute__((ext_vector_type(8)))  float  v8f;

__device__ __forceinline__ __bf16 tobf(float f) { return (__bf16)f; }
// WMMA 16-bit fragment K index for element e in lane: ISA 7.12.2 layout
__device__ __forceinline__ int kmap(int e, int lane) { return e + 8*((e>>3) + (lane>>4)); }

// ---- Phase 1: q projection (16x1024 @ 1024x1024, tiny) ----
__global__ void k_qproj(const float* __restrict__ query, const float* __restrict__ Wq,
                        float* __restrict__ qproj) {
  int idx = blockIdx.x*256 + threadIdx.x;          // 16384 outputs
  int m = idx & (DM-1); int b = idx >> 10;
  float s = 0.f;
  for (int k = 0; k < DM; ++k) s += query[b*DM + k] * Wq[k*DM + m];
  qproj[idx] = s;
}

// ---- Phase 1b: fold Wk into (q+qpb):  A[b,h,m] = sum_d Wk[m,h*64+d]*(q+qpb)[b,h,d] ----
__global__ void k_aw(const float* __restrict__ qproj, const float* __restrict__ qpb,
                     const float* __restrict__ Wk, float* __restrict__ Aw) {
  int idx = blockIdx.x*256 + threadIdx.x;          // 262144 outputs, [b][h][m]
  int m = idx & (DM-1); int hb = idx >> 10; int h = hb & 15; int b = hb >> 4;
  float s = 0.f;
  #pragma unroll 8
  for (int d = 0; d < DK; ++d)
    s += Wk[m*DM + h*DK + d] * (qproj[b*DM + h*DK + d] + qpb[h*DK + d]);
  Aw[idx] = s;
}

// ---- Phase 1c: positional term bd[j,b,h] = q[b,h,:].kpe[j,h,:] + kpb[j,h] ----
__global__ void k_bd(const float* __restrict__ qproj, const float* __restrict__ kpe,
                     const float* __restrict__ kpb, float* __restrict__ bdterm) {
  int j = blockIdx.x;                              // 4096
  int t = threadIdx.x;                             // 256 = b*16+h
  int b = t >> 4, h = t & 15;
  float s = 0.f;
  #pragma unroll 8
  for (int d = 0; d < DK; ++d)
    s += qproj[b*DM + h*DK + d] * kpe[j*DM + h*DK + d];
  bdterm[(size_t)j*BH + t] = s + kpb[j*H_ + h];
}

// ---- Phase 2 (HBM stream #1): scores[j,b,h] via WMMA key_tile(16x32) @ A(32x16) ----
__global__ void k_scores(const float* __restrict__ key, const float* __restrict__ Aw,
                         const float* __restrict__ bdterm, float* __restrict__ scores) {
  int b    = blockIdx.y;
  int wave = threadIdx.x >> 5;
  int lane = threadIdx.x & 31;
  int l15  = lane & 15;
  int sub  = (lane >> 4) << 3;                     // 0 or 8: K-half per half-wave
  int jbase = blockIdx.x*128 + wave*16;

  const float* krow = key + (size_t)(jbase + l15)*BD + (size_t)b*DM;  // A: M=j row
  const float* arow = Aw  + (size_t)(b*H_ + l15)*DM;                  // B: N=h col

  v8f acc = {};
  for (int kc = 0; kc < DM; kc += 32) {
    const float* kp = krow + kc + sub;
    float4 f0 = *(const float4*)(kp);
    float4 f1 = *(const float4*)(kp + 4);
    float4 f2 = *(const float4*)(kp + 16);
    float4 f3 = *(const float4*)(kp + 20);
    v16bf a;
    a[0]=tobf(f0.x);  a[1]=tobf(f0.y);  a[2]=tobf(f0.z);  a[3]=tobf(f0.w);
    a[4]=tobf(f1.x);  a[5]=tobf(f1.y);  a[6]=tobf(f1.z);  a[7]=tobf(f1.w);
    a[8]=tobf(f2.x);  a[9]=tobf(f2.y);  a[10]=tobf(f2.z); a[11]=tobf(f2.w);
    a[12]=tobf(f3.x); a[13]=tobf(f3.y); a[14]=tobf(f3.z); a[15]=tobf(f3.w);
    const float* ap = arow + kc + sub;
    float4 g0 = *(const float4*)(ap);
    float4 g1 = *(const float4*)(ap + 4);
    float4 g2 = *(const float4*)(ap + 16);
    float4 g3 = *(const float4*)(ap + 20);
    v16bf bb;
    bb[0]=tobf(g0.x);  bb[1]=tobf(g0.y);  bb[2]=tobf(g0.z);  bb[3]=tobf(g0.w);
    bb[4]=tobf(g1.x);  bb[5]=tobf(g1.y);  bb[6]=tobf(g1.z);  bb[7]=tobf(g1.w);
    bb[8]=tobf(g2.x);  bb[9]=tobf(g2.y);  bb[10]=tobf(g2.z); bb[11]=tobf(g2.w);
    bb[12]=tobf(g3.x); bb[13]=tobf(g3.y); bb[14]=tobf(g3.z); bb[15]=tobf(g3.w);
    acc = __builtin_amdgcn_wmma_f32_16x16x32_bf16(false, a, false, bb, (short)0, acc,
                                                  false, false);
  }
  const float scale = 0.125f;                       // 1/sqrt(64)
  int joff = jbase + sub;                           // D-matrix: M = r + 8*(lane>=16)
  #pragma unroll
  for (int r = 0; r < 8; ++r) {
    size_t o = (size_t)(joff + r)*BH + b*H_ + l15;
    scores[o] = (acc[r] + bdterm[o]) * scale;
  }
}

// ---- Phase 3: softmax over j per (b,h); in-place -> attn ----
__global__ void k_softmax(float* __restrict__ scores) {
  int bh = blockIdx.x;                              // 256
  int t  = threadIdx.x;                             // 256
  __shared__ float red[256];
  float vals[16];
  float lmax = -3.4e38f;
  #pragma unroll
  for (int i = 0; i < 16; ++i) {
    vals[i] = scores[(size_t)(t + i*256)*BH + bh];
    lmax = fmaxf(lmax, vals[i]);
  }
  red[t] = lmax; __syncthreads();
  for (int s = 128; s > 0; s >>= 1) {
    if (t < s) red[t] = fmaxf(red[t], red[t+s]);
    __syncthreads();
  }
  float bmax = red[0]; __syncthreads();
  float lsum = 0.f;
  #pragma unroll
  for (int i = 0; i < 16; ++i) { vals[i] = __expf(vals[i] - bmax); lsum += vals[i]; }
  red[t] = lsum; __syncthreads();
  for (int s = 128; s > 0; s >>= 1) {
    if (t < s) red[t] += red[t+s];
    __syncthreads();
  }
  float inv = 1.0f / red[0];
  #pragma unroll
  for (int i = 0; i < 16; ++i)
    scores[(size_t)(t + i*256)*BH + bh] = vals[i] * inv;
}

// ---- Phase 4 (HBM stream #2): y[b,h,m] via WMMA attn(16x32) @ value(32x16).
//      value/attn tiles staged in LDS through the CDNA5 async LDS-copy path
//      (global_load_async_to_lds_*, tracked on ASYNCcnt) for coalesced b128
//      streaming of the dominant 256 MB read. ----
__global__ void k_y(const float* __restrict__ attn, const float* __restrict__ value,
                    float* __restrict__ y) {
  int b    = blockIdx.y;
  int tid  = threadIdx.x;
  int lane = tid & 31;
  int wave = tid >> 5;
  int l15  = lane & 15;
  int nblk = blockIdx.x*128;

  __shared__ float att[32*16];     // [32 j][16 h] attn tile (shared by all 8 waves)
  __shared__ float vt[32*128];     // [32 j][128 m] value tile

  // attn tile: 512 floats / 256 threads = one async b64 each
  int aj = tid >> 3, ap = (tid & 7) * 2;
  const float* asrc = attn + (size_t)aj*BH + b*H_ + ap;
  unsigned adst = (unsigned)(uintptr_t)&att[aj*16 + ap];

  const float* vbase = value + (size_t)b*DM + nblk;

  v8f acc = {};
  for (int kc = 0; kc < S_; kc += 32) {
    {
      unsigned long long ga = (unsigned long long)(uintptr_t)(asrc + (size_t)kc*BH);
      asm volatile("global_load_async_to_lds_b64 %0, %1, off"
                   :: "v"(adst), "v"(ga) : "memory");
    }
    #pragma unroll
    for (int q = 0; q < 4; ++q) {                   // value tile: 4096 floats, b128 each
      int idx = q*256 + tid;                        // 0..1023
      int j = idx >> 5, c4 = (idx & 31) * 4;
      unsigned long long ga =
          (unsigned long long)(uintptr_t)(vbase + (size_t)(kc + j)*BD + c4);
      unsigned ld = (unsigned)(uintptr_t)&vt[j*128 + c4];
      asm volatile("global_load_async_to_lds_b128 %0, %1, off"
                   :: "v"(ld), "v"(ga) : "memory");
    }
    asm volatile("s_wait_asynccnt 0" ::: "memory"); // my async writes landed in LDS
    __syncthreads();                                // everyone's landed
    v16bf a, bb;
    #pragma unroll
    for (int e = 0; e < 16; ++e) {
      int km = kmap(e, lane);
      a[e]  = tobf(att[km*16 + l15]);               // A: M=h rows of attn^T
      bb[e] = tobf(vt[km*128 + wave*16 + l15]);     // B: K=j, N=m of value
    }
    acc = __builtin_amdgcn_wmma_f32_16x16x32_bf16(false, a, false, bb, (short)0, acc,
                                                  false, false);
    __syncthreads();                                // reads done before next overwrite
  }
  int hbase = (lane >> 4) << 3;
  #pragma unroll
  for (int r = 0; r < 8; ++r)
    y[(size_t)(b*H_ + hbase + r)*DM + nblk + wave*16 + l15] = acc[r];
}

// ---- Phase 5a: x[b,h,d] = sum_m y[b,h,m]*Wv[m,h*64+d] + bv (softmax sums to 1) ----
__global__ void k_x(const float* __restrict__ y, const float* __restrict__ Wv,
                    const float* __restrict__ bv, float* __restrict__ x) {
  int idx = blockIdx.x*256 + threadIdx.x;           // 16384, [b][h][d]
  int d = idx & 63; int h = (idx>>6) & 15; int b = idx >> 10;
  float s = 0.f;
  for (int m = 0; m < DM; ++m)
    s += y[(size_t)(b*H_ + h)*DM + m] * Wv[m*DM + h*DK + d];
  x[idx] = s + bv[h*DK + d];
}

// ---- Phase 5b: out = x @ Wo + bo ----
__global__ void k_out(const float* __restrict__ x, const float* __restrict__ Wo,
                      const float* __restrict__ bo, float* __restrict__ out) {
  int idx = blockIdx.x*256 + threadIdx.x;           // 16384
  int n = idx & (DM-1); int b = idx >> 10;
  float s = 0.f;
  for (int t = 0; t < DM; ++t) s += x[b*DM + t] * Wo[t*DM + n];
  out[idx] = s + bo[n];
}

extern "C" void kernel_launch(void* const* d_in, const int* in_sizes, int n_in,
                              void* d_out, int out_size, void* d_ws, size_t ws_size,
                              hipStream_t stream) {
  const float* query = (const float*)d_in[0];
  const float* key   = (const float*)d_in[1];
  const float* value = (const float*)d_in[2];
  const float* Wq    = (const float*)d_in[3];
  const float* Wk    = (const float*)d_in[4];
  const float* Wv    = (const float*)d_in[5];
  const float* bv    = (const float*)d_in[6];
  const float* Wo    = (const float*)d_in[7];
  const float* bo    = (const float*)d_in[8];
  const float* kpe   = (const float*)d_in[9];
  const float* kpb   = (const float*)d_in[10];
  const float* qpb   = (const float*)d_in[11];
  float* out = (float*)d_out;

  float* ws     = (float*)d_ws;
  float* qproj  = ws;                    // 16384
  float* Aw     = qproj  + 16384;        // 262144
  float* bdterm = Aw     + 262144;       // 1048576
  float* scores = bdterm + 1048576;      // 1048576 (softmax in place -> attn)
  float* y      = scores + 1048576;      // 262144
  float* x      = y      + 262144;       // 16384  (total ~10.6 MB)

  k_qproj  <<<dim3(64),    dim3(256), 0, stream>>>(query, Wq, qproj);
  k_aw     <<<dim3(1024),  dim3(256), 0, stream>>>(qproj, qpb, Wk, Aw);
  k_bd     <<<dim3(4096),  dim3(256), 0, stream>>>(qproj, kpe, kpb, bdterm);
  k_scores <<<dim3(32,16), dim3(256), 0, stream>>>(key, Aw, bdterm, scores);
  k_softmax<<<dim3(256),   dim3(256), 0, stream>>>(scores);
  k_y      <<<dim3(8,16),  dim3(256), 0, stream>>>(scores, value, y);
  k_x      <<<dim3(64),    dim3(256), 0, stream>>>(y, Wv, bv, x);
  k_out    <<<dim3(64),    dim3(256), 0, stream>>>(x, Wo, bo, out);
}
